// MultiHeadSelfAttention_55018531062301
// MI455X (gfx1250) — compile-verified, matrix operand order checked by reading, not run on hardware
//
#include <hip/hip_runtime.h>

// ---------------------------------------------------------------------------
// MHSA + RoPE for MI455X (gfx1250, wave32).
// All GEMMs via v_wmma_f32_16x16x32_f16 (f16 inputs, f32 accumulate).
// Attention K/V tiles staged into LDS by the Tensor Data Mover (TDM) with
// LDS row padding for bank-conflict-free ds_load_b128 fragment loads.
// Workspace layout (f16 halves): xh | qwh | kwh | vwh | owh | Qh | Kh | Vt | Ah
// Needs 48 MB of d_ws.
// ---------------------------------------------------------------------------

typedef __attribute__((ext_vector_type(16))) _Float16 v16h;
typedef __attribute__((ext_vector_type(8)))  _Float16 v8h;
typedef __attribute__((ext_vector_type(8)))  float    v8f;
typedef __attribute__((ext_vector_type(4)))  unsigned int u32x4;
typedef __attribute__((ext_vector_type(8)))  int          i32x8;

#define D_MODEL   1024
#define NUM_HEADS 16
#define D_K       64
#define SEQ       2048
#define BATCH     2
#define BSR       (BATCH * SEQ)   // 4096 rows
#define LDSPITCH  72              // 64 halves + 4-dword TDM pad = 36 dwords

__device__ __forceinline__ v8f wmma_f16(v16h a, v16h b, v8f c) {
  return __builtin_amdgcn_wmma_f32_16x16x32_f16(false, a, false, b, (short)0, c,
                                                false, false);
}

// Load a 16x32 f16 fragment (A layout; B layout is the mirrored analog) from a
// row-major K-contiguous matrix. Per ISA 16-bit A layout: lane supplies row
// (lane&15); kbase=(lane>>4)*8; VGPR0-3 hold k=kbase..kbase+7, VGPR4-7 hold
// k=kbase+16..kbase+23  -> two 16-byte loads per lane.
__device__ __forceinline__ v16h load_frag(const _Float16* p, int ld) {
  const int lane = threadIdx.x & 31;
  const _Float16* r = p + (size_t)(lane & 15) * ld + ((lane >> 4) << 3);
  union { v16h v; v8h h[2]; } u;
  u.h[0] = *(const v8h*)(r);
  u.h[1] = *(const v8h*)(r + 16);
  return u.v;
}

// ---------------------------------------------------------------------------
// TDM: 2-D tensor tile -> LDS (async, TENSORcnt).  D# per CDNA5 ISA ch.8:
//  group0: [1:0]=count=1 | [63:32]=lds_addr | [120:64]=global_addr |
//          [127:126]=type=2
//  group1: [17:16]=data_size(1->2B) | [20]=pad_en | [24:22]=pad_interval |
//          [31:25]=pad_amount | [79:48]=tensor_dim0 | [111:80]=tensor_dim1 |
//          [127:112]=tile_dim0 | [143:128]=tile_dim1 | [207:160]=dim0_stride
//  pad: interval code 4 = 32 dwords (one 64-half row), amount code 3 = 4
//  dwords -> LDS row pitch 36 dwords (bank-conflict-free, 16B aligned).
// 2-D tensor => 2-operand asm form (VADDR2/VADDR3 NULL).
// ---------------------------------------------------------------------------
__device__ __forceinline__ void tdm_load_2d(const void* gaddr, unsigned lds_off,
                                            unsigned td0, unsigned td1,
                                            unsigned stride0, unsigned tl0,
                                            unsigned tl1) {
  const unsigned long long ga = (unsigned long long)gaddr;
  u32x4 g0;
  g0[0] = 1u;                                        // count=1 (user D#)
  g0[1] = lds_off;                                   // lds_addr (bytes)
  g0[2] = (unsigned)(ga & 0xFFFFFFFFu);              // global_addr[31:0]
  g0[3] = (unsigned)((ga >> 32) & 0x01FFFFFFu)       // global_addr[56:32]
        | (2u << 30);                                // type = 2 ("image")
  i32x8 g1;
  g1[0] = (int)((1u << 16)                           // data_size = 2 bytes
        | (1u << 20)                                 // pad_enable
        | (4u << 22)                                 // pad_interval: 32 dwords
        | (3u << 25));                               // pad_amount: 4 dwords
  g1[1] = (int)((td0 & 0xFFFFu) << 16);              // tensor_dim0[15:0]
  g1[2] = (int)((td0 >> 16) | ((td1 & 0xFFFFu) << 16));
  g1[3] = (int)((td1 >> 16) | (tl0 << 16));          // tile_dim0
  g1[4] = (int)(tl1 & 0xFFFFu);                      // tile_dim1 (tile_dim2=0)
  g1[5] = (int)stride0;                              // dim0_stride[31:0]
  g1[6] = 0;
  g1[7] = 0;
  asm volatile("tensor_load_to_lds %0, %1" :: "s"(g0), "s"(g1) : "memory");
}

// ---------------------------------------------------------------------------
// Stage 1: fp32 -> f16 conversion
// ---------------------------------------------------------------------------
__global__ void k_cvt(const float* __restrict__ in, _Float16* __restrict__ out,
                      int n) {
  int i = blockIdx.x * blockDim.x + threadIdx.x;
  const int stride = gridDim.x * blockDim.x;
  for (; i < n; i += stride) out[i] = (_Float16)in[i];
}

// ---------------------------------------------------------------------------
// Stage 2: fused QKV projection + RoPE.  grid = (M/128, N/128, 3)
// Wave computes a 32x64 tile (2x4 WMMA C tiles), WG = 8 waves = 128x128.
// z=0 -> Q (RoPE, (b,h,s,d) f16), z=1 -> K (RoPE), z=2 -> V stored transposed
// as (b,h,d,s) so attention's P*V B-fragments are K-contiguous.
// ---------------------------------------------------------------------------
__global__ __launch_bounds__(256) void k_qkv(
    const _Float16* __restrict__ xh, const _Float16* __restrict__ qwh,
    const _Float16* __restrict__ kwh, const _Float16* __restrict__ vwh,
    const int* __restrict__ tok, _Float16* __restrict__ Qh,
    _Float16* __restrict__ Kh, _Float16* __restrict__ Vt) {
  const int lane = threadIdx.x & 31;
  const int wid  = threadIdx.x >> 5;
  const int ln = lane & 15, hf = lane >> 4;
  const int m_base = blockIdx.x * 128 + (wid >> 1) * 32;
  const int n_base = blockIdx.y * 128 + (wid & 1) * 64;
  const int z = blockIdx.z;
  const _Float16* W = (z == 0) ? qwh : (z == 1) ? kwh : vwh;

  v8f acc[2][4];
#pragma unroll
  for (int i = 0; i < 2; i++)
#pragma unroll
    for (int j = 0; j < 4; j++)
#pragma unroll
      for (int r = 0; r < 8; r++) acc[i][j][r] = 0.f;

  for (int k0 = 0; k0 < D_MODEL; k0 += 32) {
    if (k0 + 32 < D_MODEL) {
      __builtin_prefetch(xh + (size_t)(m_base + ln) * D_MODEL + k0 + 32, 0, 3);
      __builtin_prefetch(W + (size_t)(n_base + ln) * D_MODEL + k0 + 32, 0, 3);
    }
    v16h a0 = load_frag(xh + (size_t)(m_base +  0) * D_MODEL + k0, D_MODEL);
    v16h a1 = load_frag(xh + (size_t)(m_base + 16) * D_MODEL + k0, D_MODEL);
    v16h b0 = load_frag(W + (size_t)(n_base +  0) * D_MODEL + k0, D_MODEL);
    v16h b1 = load_frag(W + (size_t)(n_base + 16) * D_MODEL + k0, D_MODEL);
    v16h b2 = load_frag(W + (size_t)(n_base + 32) * D_MODEL + k0, D_MODEL);
    v16h b3 = load_frag(W + (size_t)(n_base + 48) * D_MODEL + k0, D_MODEL);
    acc[0][0] = wmma_f16(a0, b0, acc[0][0]);
    acc[0][1] = wmma_f16(a0, b1, acc[0][1]);
    acc[0][2] = wmma_f16(a0, b2, acc[0][2]);
    acc[0][3] = wmma_f16(a0, b3, acc[0][3]);
    acc[1][0] = wmma_f16(a1, b0, acc[1][0]);
    acc[1][1] = wmma_f16(a1, b1, acc[1][1]);
    acc[1][2] = wmma_f16(a1, b2, acc[1][2]);
    acc[1][3] = wmma_f16(a1, b3, acc[1][3]);
  }

  // C layout: lane holds column (n_base + j*16 + ln); rows m = i*16 + hf*8 + r.
  if (z < 2) {
    _Float16* dst = (z == 0) ? Qh : Kh;
#pragma unroll
    for (int i = 0; i < 2; i++) {
      const int m0 = m_base + i * 16 + hf * 8;
      float posf[8];
#pragma unroll
      for (int r = 0; r < 8; r++) posf[r] = (float)tok[m0 + r];
#pragma unroll
      for (int j = 0; j < 4; j++) {
        const int col = n_base + j * 16 + ln;
        const int hh = col >> 6, d = col & 63;
        // inv_freq = 10000^(-(d/2)/32) ; ln(10000)/32 = 0.287823136...
        const float inv = __expf(-0.28782313662425572f * (float)(d >> 1));
        const bool even = (d & 1) == 0;
#pragma unroll
        for (int r = 0; r < 8; r++) {
          float v  = acc[i][j][r];
          float pv = __shfl_xor(v, 1, 32);  // partner column (even<->odd)
          float sn, cs;
          __sincosf(posf[r] * inv, &sn, &cs);
          float res = even ? (v * cs - pv * sn) : (pv * sn + v * cs);
          const int m = m0 + r;
          const int bb = m >> 11, s = m & (SEQ - 1);
          dst[((size_t)((bb * NUM_HEADS + hh) * SEQ + s)) * D_K + d] =
              (_Float16)res;
        }
      }
    }
  } else {
    // V transposed: Vt[(b,h,d), s]; lane's 8 rows are consecutive s ->
    // one 16B packed store per tile.
#pragma unroll
    for (int i = 0; i < 2; i++) {
      const int m0 = m_base + i * 16 + hf * 8;
      const int bb = m0 >> 11, s0 = m0 & (SEQ - 1);
#pragma unroll
      for (int j = 0; j < 4; j++) {
        const int col = n_base + j * 16 + ln;
        const int hh = col >> 6, d = col & 63;
        v8h hv;
#pragma unroll
        for (int r = 0; r < 8; r++) hv[r] = (_Float16)acc[i][j][r];
        *(v8h*)(Vt + ((size_t)((bb * NUM_HEADS + hh) * D_K + d)) * SEQ + s0) =
            hv;
      }
    }
  }
}

// ---------------------------------------------------------------------------
// Stage 3: causal flash attention. grid = (S/64, H, B), block = 128 (4 waves).
// Per key block: wave 0 TDM-loads the 64x64 K tile and V tile into LDS
// (shared by all 4 waves -> 4x less global traffic), s_wait_tensorcnt,
// barrier; each wave owns 16 query rows, online softmax in f32, P re-laid
// out C->A through per-wave LDS (wave-internal, in-order DS).
// ---------------------------------------------------------------------------
__global__ __launch_bounds__(128) void k_attn(const _Float16* __restrict__ Qh,
                                              const _Float16* __restrict__ Kh,
                                              const _Float16* __restrict__ Vt,
                                              _Float16* __restrict__ Ah) {
  __shared__ __align__(16) _Float16 Kl[64 * LDSPITCH];   // K tile, padded rows
  __shared__ __align__(16) _Float16 Vl[64 * LDSPITCH];   // V tile, padded rows
  __shared__ __align__(16) _Float16 pt[4][16 * LDSPITCH];
  const int lane = threadIdx.x & 31;
  const int wid  = threadIdx.x >> 5;
  const int ln = lane & 15, hf = lane >> 4;
  const int b = blockIdx.z, h = blockIdx.y;
  const int qblk = blockIdx.x * 64;
  const int q0 = qblk + wid * 16;

  const _Float16* Qb = Qh + ((size_t)(b * NUM_HEADS + h)) * SEQ * D_K;
  const _Float16* Kb = Kh + ((size_t)(b * NUM_HEADS + h)) * SEQ * D_K;
  const _Float16* Vb = Vt + ((size_t)(b * NUM_HEADS + h)) * D_K * SEQ;
  _Float16* pw = &pt[wid][0];

  const v16h aq0 = load_frag(Qb + (size_t)q0 * D_K, D_K);
  const v16h aq1 = load_frag(Qb + (size_t)q0 * D_K + 32, D_K);

  v8f m_run, l_run, accd[4];
#pragma unroll
  for (int r = 0; r < 8; r++) { m_run[r] = -1e30f; l_run[r] = 0.f; }
#pragma unroll
  for (int t = 0; t < 4; t++)
#pragma unroll
    for (int r = 0; r < 8; r++) accd[t][r] = 0.f;

  for (int kb = 0; kb <= qblk; kb += 64) {
    if (wid == 0) {
      // K tile: tensor (SEQ x 64) halves, tile 64 rows x 64; tile start row kb
      tdm_load_2d(Kb + (size_t)kb * D_K, (unsigned)(size_t)(void*)Kl,
                  D_K, SEQ, D_K, D_K, 64);
      // V tile: tensor (64 x SEQ) halves (d-major), tile 64 x 64 at col kb
      tdm_load_2d(Vb + kb, (unsigned)(size_t)(void*)Vl,
                  SEQ, D_K, SEQ, 64, D_K);
      __builtin_amdgcn_s_wait_tensorcnt(0);
    }
    __syncthreads();

    v8f p[4], bmax;
#pragma unroll
    for (int r = 0; r < 8; r++) bmax[r] = -1e30f;

#pragma unroll
    for (int j = 0; j < 4; j++) {
      v16h bk0 = load_frag(Kl + (j * 16) * LDSPITCH, LDSPITCH);
      v16h bk1 = load_frag(Kl + (j * 16) * LDSPITCH + 32, LDSPITCH);
      v8f s;
#pragma unroll
      for (int r = 0; r < 8; r++) s[r] = 0.f;
      s = wmma_f16(aq0, bk0, s);
      s = wmma_f16(aq1, bk1, s);
      const int kg = kb + j * 16 + ln;
#pragma unroll
      for (int r = 0; r < 8; r++) {
        const int qg = q0 + hf * 8 + r;
        float v = s[r] * 0.125f;             // 1/sqrt(64)
        v = (kg <= qg) ? v : -1e30f;         // causal mask
        s[r] = v;
        bmax[r] = fmaxf(bmax[r], v);
      }
      p[j] = s;
    }

    // row max over the 16 lanes of each half-wave
#pragma unroll
    for (int mm = 1; mm < 16; mm <<= 1)
#pragma unroll
      for (int r = 0; r < 8; r++)
        bmax[r] = fmaxf(bmax[r], __shfl_xor(bmax[r], mm, 32));

    v8f scale, rs;
#pragma unroll
    for (int r = 0; r < 8; r++) {
      float mn = fmaxf(m_run[r], bmax[r]);
      scale[r] = __expf(m_run[r] - mn);
      m_run[r] = mn;
      rs[r] = 0.f;
    }
#pragma unroll
    for (int j = 0; j < 4; j++)
#pragma unroll
      for (int r = 0; r < 8; r++) {
        float e = __expf(p[j][r] - m_run[r]);
        p[j][r] = e;
        rs[r] += e;
      }
#pragma unroll
    for (int mm = 1; mm < 16; mm <<= 1)
#pragma unroll
      for (int r = 0; r < 8; r++) rs[r] += __shfl_xor(rs[r], mm, 32);
#pragma unroll
    for (int r = 0; r < 8; r++) l_run[r] = l_run[r] * scale[r] + rs[r];
#pragma unroll
    for (int t = 0; t < 4; t++)
#pragma unroll
      for (int r = 0; r < 8; r++) accd[t][r] *= scale[r];

    // P (f32, C layout) -> LDS f16 16x64 tile (padded pitch), reload A layout.
#pragma unroll
    for (int j = 0; j < 4; j++)
#pragma unroll
      for (int r = 0; r < 8; r++)
        pw[(hf * 8 + r) * LDSPITCH + j * 16 + ln] = (_Float16)p[j][r];
    v16h ap0 = load_frag(pw, LDSPITCH);
    v16h ap1 = load_frag(pw + 32, LDSPITCH);

#pragma unroll
    for (int t = 0; t < 4; t++) {
      v16h bv0 = load_frag(Vl + (t * 16) * LDSPITCH, LDSPITCH);
      v16h bv1 = load_frag(Vl + (t * 16) * LDSPITCH + 32, LDSPITCH);
      accd[t] = wmma_f16(ap0, bv0, accd[t]);
      accd[t] = wmma_f16(ap1, bv1, accd[t]);
    }
    __syncthreads();   // protect K/V LDS tiles before next TDM overwrite
  }

  // normalize and store to Ah (b, s, h*64+d) f16
#pragma unroll
  for (int t = 0; t < 4; t++) {
    const int col = h * D_K + t * 16 + ln;
#pragma unroll
    for (int r = 0; r < 8; r++) {
      const int s = q0 + hf * 8 + r;
      float o = accd[t][r] / l_run[r];
      Ah[((size_t)(b * SEQ + s)) * D_MODEL + col] = (_Float16)o;
    }
  }
}

// ---------------------------------------------------------------------------
// Stage 4: output projection  out = A * o_w^T  (fp32 result)
// ---------------------------------------------------------------------------
__global__ __launch_bounds__(256) void k_oproj(const _Float16* __restrict__ Ahh,
                                               const _Float16* __restrict__ Wh,
                                               float* __restrict__ out) {
  const int lane = threadIdx.x & 31;
  const int wid  = threadIdx.x >> 5;
  const int ln = lane & 15, hf = lane >> 4;
  const int m_base = blockIdx.x * 128 + (wid >> 1) * 32;
  const int n_base = blockIdx.y * 128 + (wid & 1) * 64;

  v8f acc[2][4];
#pragma unroll
  for (int i = 0; i < 2; i++)
#pragma unroll
    for (int j = 0; j < 4; j++)
#pragma unroll
      for (int r = 0; r < 8; r++) acc[i][j][r] = 0.f;

  for (int k0 = 0; k0 < D_MODEL; k0 += 32) {
    if (k0 + 32 < D_MODEL) {
      __builtin_prefetch(Ahh + (size_t)(m_base + ln) * D_MODEL + k0 + 32, 0, 3);
      __builtin_prefetch(Wh + (size_t)(n_base + ln) * D_MODEL + k0 + 32, 0, 3);
    }
    v16h a0 = load_frag(Ahh + (size_t)(m_base +  0) * D_MODEL + k0, D_MODEL);
    v16h a1 = load_frag(Ahh + (size_t)(m_base + 16) * D_MODEL + k0, D_MODEL);
    v16h b0 = load_frag(Wh + (size_t)(n_base +  0) * D_MODEL + k0, D_MODEL);
    v16h b1 = load_frag(Wh + (size_t)(n_base + 16) * D_MODEL + k0, D_MODEL);
    v16h b2 = load_frag(Wh + (size_t)(n_base + 32) * D_MODEL + k0, D_MODEL);
    v16h b3 = load_frag(Wh + (size_t)(n_base + 48) * D_MODEL + k0, D_MODEL);
    acc[0][0] = wmma_f16(a0, b0, acc[0][0]);
    acc[0][1] = wmma_f16(a0, b1, acc[0][1]);
    acc[0][2] = wmma_f16(a0, b2, acc[0][2]);
    acc[0][3] = wmma_f16(a0, b3, acc[0][3]);
    acc[1][0] = wmma_f16(a1, b0, acc[1][0]);
    acc[1][1] = wmma_f16(a1, b1, acc[1][1]);
    acc[1][2] = wmma_f16(a1, b2, acc[1][2]);
    acc[1][3] = wmma_f16(a1, b3, acc[1][3]);
  }

#pragma unroll
  for (int i = 0; i < 2; i++) {
    const int m0 = m_base + i * 16 + hf * 8;
#pragma unroll
    for (int j = 0; j < 4; j++) {
      const int col = n_base + j * 16 + ln;
#pragma unroll
      for (int r = 0; r < 8; r++)
        out[(size_t)(m0 + r) * D_MODEL + col] = acc[i][j][r];
    }
  }
}

// ---------------------------------------------------------------------------
extern "C" void kernel_launch(void* const* d_in, const int* in_sizes, int n_in,
                              void* d_out, int out_size, void* d_ws,
                              size_t ws_size, hipStream_t stream) {
  (void)in_sizes; (void)n_in; (void)out_size; (void)ws_size;
  const float* x  = (const float*)d_in[0];
  const float* qw = (const float*)d_in[1];
  const float* kw = (const float*)d_in[2];
  const float* vw = (const float*)d_in[3];
  const float* ow = (const float*)d_in[4];
  const int* tok  = (const int*)d_in[5];
  float* out = (float*)d_out;

  const size_t NX = (size_t)BSR * D_MODEL;        // 4M halves
  const size_t NW = (size_t)D_MODEL * D_MODEL;    // 1M halves
  _Float16* xh  = (_Float16*)d_ws;
  _Float16* qwh = xh  + NX;
  _Float16* kwh = qwh + NW;
  _Float16* vwh = kwh + NW;
  _Float16* owh = vwh + NW;
  _Float16* Qh  = owh + NW;
  _Float16* Kh  = Qh + NX;
  _Float16* Vth = Kh + NX;
  _Float16* Ahh = Vth + NX;   // total 24M halves = 48 MB

  k_cvt<<<dim3(2048), dim3(256), 0, stream>>>(x,  xh,  (int)NX);
  k_cvt<<<dim3(512),  dim3(256), 0, stream>>>(qw, qwh, (int)NW);
  k_cvt<<<dim3(512),  dim3(256), 0, stream>>>(kw, kwh, (int)NW);
  k_cvt<<<dim3(512),  dim3(256), 0, stream>>>(vw, vwh, (int)NW);
  k_cvt<<<dim3(512),  dim3(256), 0, stream>>>(ow, owh, (int)NW);

  k_qkv<<<dim3(BSR / 128, D_MODEL / 128, 3), dim3(256), 0, stream>>>(
      xh, qwh, kwh, vwh, tok, Qh, Kh, Vth);

  k_attn<<<dim3(SEQ / 64, NUM_HEADS, BATCH), dim3(128), 0, stream>>>(
      Qh, Kh, Vth, Ahh);

  k_oproj<<<dim3(BSR / 128, D_MODEL / 128), dim3(256), 0, stream>>>(
      Ahh, owh, out);
}
